// SparseSelfAttention_1082331758767
// MI455X (gfx1250) — matrix operand order repaired and checked by reading, not compiled, not run on hardware
//
#include <hip/hip_runtime.h>
#include <hip/hip_bf16.h>

typedef __attribute__((ext_vector_type(16))) __bf16 v16bf;
typedef __attribute__((ext_vector_type(8)))  __bf16 v8bf;
typedef __attribute__((ext_vector_type(8)))  float  v8f;

constexpr int B_   = 2;
constexpr int S_   = 8192;
constexpr int HID_ = 768;
constexpr int NH_  = 12;
constexpr int HD_  = 64;
constexpr float SCALE_ = 0.125f; // 1/sqrt(64)

__device__ __forceinline__ v8f wmma_bf16(v16bf a, v16bf b, v8f c) {
  return __builtin_amdgcn_wmma_f32_16x16x32_bf16(false, a, false, b, (short)0, c, false, false);
}

// ---------------------------------------------------------------- converts
__global__ void cvt_f32_bf16(const float* __restrict__ src, __bf16* __restrict__ dst, long n) {
  long i = (long)blockIdx.x * blockDim.x + threadIdx.x;
  long stride = (long)gridDim.x * blockDim.x;
  for (; i < n; i += stride) dst[i] = (__bf16)src[i];
}

__global__ void zero_bf16(__bf16* __restrict__ dst, long n) {
  long i = (long)blockIdx.x * blockDim.x + threadIdx.x;
  long stride = (long)gridDim.x * blockDim.x;
  for (; i < n; i += stride) dst[i] = (__bf16)0.0f;
}

// ---------------------------------------------------------------- GEMM: Y = X @ W^T + bias
// mode 0: out f32 row-major [i*768+j]           (final output projection)
// mode 1: out bf16 head layout [bh][s][d]       (K)
// mode 2: out bf16 transposed head [bh][d][s]   (V)
// mode 3: out bf16 Qsel [bh][t][d], A-rows gathered via idx (i = b*nselpad + t)
__global__ void gemm_bf16(const __bf16* __restrict__ A, const __bf16* __restrict__ W,
                          const float* __restrict__ bias,
                          const int* __restrict__ gather, int nsel, int nselpad,
                          int Mrows, int mode,
                          __bf16* __restrict__ outb, float* __restrict__ outf) {
  const int lane = threadIdx.x & 31;
  const int wid  = threadIdx.x >> 5;
  const int mtiles = Mrows >> 4;
  const int ntiles = HID_ / 64;                    // wave covers 16 x 64 of output
  int tile = blockIdx.x * (blockDim.x >> 5) + wid;
  if (tile >= mtiles * ntiles) return;
  const int mt = tile / ntiles, nt = tile % ntiles;
  const int m0 = mt << 4, j0 = nt * 64;

  const int arow = m0 + (lane & 15);
  long srcRow;
  if (mode == 3) {
    int b = arow / nselpad, t = arow % nselpad;
    int ti = (t < nsel) ? t : (nsel - 1);
    srcRow = (long)b * S_ + gather[ti];
  } else {
    srcRow = arow;
  }
  const __bf16* __restrict__ Arow = A + srcRow * HID_;
  const int kbase = (lane >> 4) * 16;

  v8f acc[4] = {};
  for (int k0 = 0; k0 < HID_; k0 += 32) {
    union { v16bf v; v8bf h[2]; } af;
    af.h[0] = *(const v8bf*)(Arow + k0 + kbase);
    af.h[1] = *(const v8bf*)(Arow + k0 + 8 + kbase);
    const int kb = k0 + (lane >> 4) * 16;
#pragma unroll
    for (int q = 0; q < 4; ++q) {
      const __bf16* Wrow = W + (long)(j0 + q * 16 + (lane & 15)) * HID_ + kb;
      v16bf bf = *(const v16bf*)Wrow;
      acc[q] = wmma_bf16(af.v, bf, acc[q]);
    }
  }

#pragma unroll
  for (int q = 0; q < 4; ++q) {
    const int col = j0 + q * 16 + (lane & 15);
    const float bv = bias[col];
#pragma unroll
    for (int r = 0; r < 8; ++r) {
      const int row = m0 + r + ((lane >> 4) << 3);
      const float val = acc[q][r] + bv;
      if (mode == 0) {
        outf[(long)row * HID_ + col] = val;
      } else {
        long off;
        const int h = col >> 6, d = col & 63;
        if (mode == 1) {
          int b = row >> 13, s = row & (S_ - 1);
          off = ((long)(b * NH_ + h) * S_ + s) * HD_ + d;
        } else if (mode == 2) {
          int b = row >> 13, s = row & (S_ - 1);
          off = ((long)(b * NH_ + h) * HD_ + d) * S_ + s;
        } else {
          int b = row / nselpad, t = row % nselpad;
          off = ((long)(b * NH_ + h) * nselpad + t) * HD_ + d;
        }
        outb[off] = (__bf16)val;
      }
    }
  }
}

// ---------------------------------------------------------------- attention pass 1: row max + denom
// Per-lane online softmax stats over the columns owned by each lane; single
// 4-stage cross-lane log-sum-exp merge after the S loop (no per-tile shuffles).
__global__ void attn_pass1(const __bf16* __restrict__ Qsel, const __bf16* __restrict__ Kh,
                           float* __restrict__ mbuf, float* __restrict__ lbuf,
                           int nsel, int nselpad) {
  const int lane = threadIdx.x & 31;
  const int wid  = threadIdx.x >> 5;
  const int Tt = nselpad >> 4;
  int unit = blockIdx.x * (blockDim.x >> 5) + wid;
  if (unit >= B_ * NH_ * Tt) return;
  const int bh = unit / Tt, tt = unit % Tt;
  const int t0 = tt << 4;

  const __bf16* __restrict__ Qrow = Qsel + ((long)bh * nselpad + t0 + (lane & 15)) * HD_;
  const int kbase = (lane >> 4) * 16;
  union { v16bf v; v8bf h[2]; } a0, a1;
  a0.h[0] = *(const v8bf*)(Qrow + kbase);
  a0.h[1] = *(const v8bf*)(Qrow + 8 + kbase);
  a1.h[0] = *(const v8bf*)(Qrow + 32 + kbase);
  a1.h[1] = *(const v8bf*)(Qrow + 40 + kbase);

  const __bf16* __restrict__ Kbase = Kh + (long)bh * S_ * HD_;
  const int kb = (lane >> 4) * 16;

  float mrun[8], lrun[8];
#pragma unroll
  for (int r = 0; r < 8; ++r) { mrun[r] = -1e30f; lrun[r] = 0.f; }

  for (int n0 = 0; n0 < S_; n0 += 16) {
    const __bf16* Krow = Kbase + (long)(n0 + (lane & 15)) * HD_;
    if (n0 + 64 < S_)
      __builtin_prefetch(Kbase + (long)(n0 + 64 + (lane & 15)) * HD_, 0, 0);
    v16bf b0 = *(const v16bf*)(Krow + kb);
    v16bf b1 = *(const v16bf*)(Krow + 32 + kb);
    v8f c = {};
    c = wmma_bf16(a0.v, b0, c);
    c = wmma_bf16(a1.v, b1, c);
#pragma unroll
    for (int r = 0; r < 8; ++r) {
      const float s = c[r] * SCALE_;
      const float mnew = fmaxf(mrun[r], s);
      lrun[r] = lrun[r] * __expf(mrun[r] - mnew) + __expf(s - mnew);
      mrun[r] = mnew;
    }
  }

  // cross-lane (m,l) merge across the 16 lanes that share each row
#pragma unroll
  for (int r = 0; r < 8; ++r) {
    float m = mrun[r], l = lrun[r];
#pragma unroll
    for (int mask = 1; mask <= 8; mask <<= 1) {
      const float om = __shfl_xor(m, mask, 32);
      const float ol = __shfl_xor(l, mask, 32);
      const float mn = fmaxf(m, om);
      l = l * __expf(m - mn) + ol * __expf(om - mn);
      m = mn;
    }
    mrun[r] = m; lrun[r] = l;
  }

  if ((lane & 15) == 0) {
    const int rb = (lane >> 4) << 3;
#pragma unroll
    for (int r = 0; r < 8; ++r) {
      mbuf[(long)bh * nselpad + t0 + rb + r] = mrun[r];
      lbuf[(long)bh * nselpad + t0 + rb + r] = lrun[r];
    }
  }
}

// ---------------------------------------------------------------- attention pass 2: probs + ctx
__global__ void attn_pass2(const __bf16* __restrict__ Qsel, const __bf16* __restrict__ Kh,
                           const __bf16* __restrict__ Vt,
                           const float* __restrict__ mbuf, const float* __restrict__ lbuf,
                           const int* __restrict__ idx,
                           float* __restrict__ probs, __bf16* __restrict__ ctxb,
                           int nsel, int nselpad) {
  __shared__ __bf16 lds[4][16 * 32];   // per-wave 16x32 bf16 P tile
  const int lane = threadIdx.x & 31;
  const int wid  = threadIdx.x >> 5;
  const int Tt = nselpad >> 4;
  int unit = blockIdx.x * (blockDim.x >> 5) + wid;
  if (unit >= B_ * NH_ * Tt) return;
  const int bh = unit / Tt, tt = unit % Tt;
  const int t0 = tt << 4;
  const int b = bh / NH_, h = bh % NH_;

  const __bf16* __restrict__ Qrow = Qsel + ((long)bh * nselpad + t0 + (lane & 15)) * HD_;
  const int kbase = (lane >> 4) * 16;
  union { v16bf v; v8bf h[2]; } a0, a1;
  a0.h[0] = *(const v8bf*)(Qrow + kbase);
  a0.h[1] = *(const v8bf*)(Qrow + 8 + kbase);
  a1.h[0] = *(const v8bf*)(Qrow + 32 + kbase);
  a1.h[1] = *(const v8bf*)(Qrow + 40 + kbase);

  const __bf16* __restrict__ Kbase = Kh + (long)bh * S_ * HD_;
  const __bf16* __restrict__ Vbase = Vt + (long)bh * HD_ * S_;
  const int kb = (lane >> 4) * 16;
  const int rb = (lane >> 4) << 3;

  float mrow[8], linv[8];
#pragma unroll
  for (int r = 0; r < 8; ++r) {
    mrow[r] = mbuf[(long)bh * nselpad + t0 + rb + r];
    linv[r] = 1.0f / lbuf[(long)bh * nselpad + t0 + rb + r];
  }

  v8f acc[4] = {};
  for (int n0 = 0; n0 < S_; n0 += 32) {
#pragma unroll
    for (int half = 0; half < 2; ++half) {
      const int nb = n0 + half * 16;
      const __bf16* Krow = Kbase + (long)(nb + (lane & 15)) * HD_;
      v16bf b0 = *(const v16bf*)(Krow + kb);
      v16bf b1 = *(const v16bf*)(Krow + 32 + kb);
      v8f c = {};
      c = wmma_bf16(a0.v, b0, c);
      c = wmma_bf16(a1.v, b1, c);
#pragma unroll
      for (int r = 0; r < 8; ++r) {
        const float p = __expf(c[r] * SCALE_ - mrow[r]) * linv[r];
        const int t = t0 + rb + r;
        const int scol = nb + (lane & 15);
        if (t < nsel) probs[((long)bh * nsel + t) * S_ + scol] = p;
        lds[wid][(rb + r) * 32 + half * 16 + (lane & 15)] = (__bf16)p;
      }
    }
    asm volatile("s_wait_dscnt 0" ::: "memory");   // intra-wave LDS RAW fence
    union { v16bf v; v8bf hh[2]; } ap;
    ap.hh[0] = *(const v8bf*)(&lds[wid][(lane & 15) * 32 + kbase]);
    ap.hh[1] = *(const v8bf*)(&lds[wid][(lane & 15) * 32 + 8 + kbase]);
#pragma unroll
    for (int q = 0; q < 4; ++q) {
      const __bf16* Vrow = Vbase + (long)(q * 16 + (lane & 15)) * S_ + n0 + ((lane >> 4) * 16);
      v16bf bv = *(const v16bf*)Vrow;
      acc[q] = wmma_bf16(ap.v, bv, acc[q]);
    }
  }

  // scatter ctx rows into bf16 context buffer (pre-zeroed)
#pragma unroll
  for (int q = 0; q < 4; ++q) {
#pragma unroll
    for (int r = 0; r < 8; ++r) {
      const int t = t0 + rb + r;
      if (t < nsel) {
        const long row = (long)b * S_ + idx[t];
        const int col = h * 64 + q * 16 + (lane & 15);
        ctxb[row * HID_ + col] = (__bf16)acc[q][r];
      }
    }
  }
}

// ---------------------------------------------------------------- launch
extern "C" void kernel_launch(void* const* d_in, const int* in_sizes, int n_in,
                              void* d_out, int out_size, void* d_ws, size_t ws_size,
                              hipStream_t stream) {
  const float* hidden = (const float*)d_in[0];
  const int*   idx    = (const int*)d_in[1];
  const float* Wq = (const float*)d_in[2]; const float* bq = (const float*)d_in[3];
  const float* Wk = (const float*)d_in[4]; const float* bk = (const float*)d_in[5];
  const float* Wv = (const float*)d_in[6]; const float* bv = (const float*)d_in[7];
  const float* Wo = (const float*)d_in[8]; const float* bo = (const float*)d_in[9];

  const int NSEL = in_sizes[1];
  const int NSELPAD = (NSEL + 15) & ~15;
  const long NROWS = (long)B_ * S_;        // 16384
  const long NHID  = NROWS * HID_;         // 12.58M
  const long WN    = (long)HID_ * HID_;    // 589824

  char* ws = (char*)d_ws;
  size_t off = 0;
  auto alloc = [&](size_t bytes) { void* p = ws + off; off += (bytes + 255) & ~(size_t)255; return p; };

  __bf16* hb   = (__bf16*)alloc(NHID * 2);
  __bf16* wqb  = (__bf16*)alloc(WN * 2);
  __bf16* wkb  = (__bf16*)alloc(WN * 2);
  __bf16* wvb  = (__bf16*)alloc(WN * 2);
  __bf16* wob  = (__bf16*)alloc(WN * 2);
  __bf16* qselb= (__bf16*)alloc((size_t)B_ * NH_ * NSELPAD * HD_ * 2);
  __bf16* khb  = (__bf16*)alloc((size_t)B_ * NH_ * S_ * HD_ * 2);
  __bf16* vtb  = (__bf16*)alloc((size_t)B_ * NH_ * S_ * HD_ * 2);
  __bf16* ctxb = (__bf16*)alloc(NHID * 2);
  float*  mbuf = (float*)alloc((size_t)B_ * NH_ * NSELPAD * 4);
  float*  lbuf = (float*)alloc((size_t)B_ * NH_ * NSELPAD * 4);

  float* out_f   = (float*)d_out;
  float* probs_f = out_f + NHID;

  // 1) fp32 -> bf16 converts; zero context
  cvt_f32_bf16<<<2048, 256, 0, stream>>>(hidden, hb, NHID);
  cvt_f32_bf16<<<512, 256, 0, stream>>>(Wq, wqb, WN);
  cvt_f32_bf16<<<512, 256, 0, stream>>>(Wk, wkb, WN);
  cvt_f32_bf16<<<512, 256, 0, stream>>>(Wv, wvb, WN);
  cvt_f32_bf16<<<512, 256, 0, stream>>>(Wo, wob, WN);
  zero_bf16<<<2048, 256, 0, stream>>>(ctxb, NHID);

  // 2) projections
  {
    int MrowsQ = B_ * NSELPAD;
    int tilesQ = (MrowsQ >> 4) * (HID_ / 64);
    gemm_bf16<<<(tilesQ + 7) / 8, 256, 0, stream>>>(hb, wqb, bq, idx, NSEL, NSELPAD,
                                                    MrowsQ, 3, qselb, nullptr);
    int tilesKV = ((int)NROWS >> 4) * (HID_ / 64);
    gemm_bf16<<<(tilesKV + 7) / 8, 256, 0, stream>>>(hb, wkb, bk, nullptr, NSEL, NSELPAD,
                                                     (int)NROWS, 1, khb, nullptr);
    gemm_bf16<<<(tilesKV + 7) / 8, 256, 0, stream>>>(hb, wvb, bv, nullptr, NSEL, NSELPAD,
                                                     (int)NROWS, 2, vtb, nullptr);
  }

  // 3) attention: pass 1 (stats), pass 2 (probs + ctx)
  {
    int units = B_ * NH_ * (NSELPAD >> 4);
    attn_pass1<<<(units + 3) / 4, 128, 0, stream>>>(qselb, khb, mbuf, lbuf, NSEL, NSELPAD);
    attn_pass2<<<(units + 3) / 4, 128, 0, stream>>>(qselb, khb, vtb, mbuf, lbuf, idx,
                                                    probs_f, ctxb, NSEL, NSELPAD);
  }

  // 4) output projection (zero rows -> bias, as in reference)
  {
    int tiles = ((int)NROWS >> 4) * (HID_ / 64);
    gemm_bf16<<<(tiles + 7) / 8, 256, 0, stream>>>(ctxb, wob, bo, nullptr, NSEL, NSELPAD,
                                                   (int)NROWS, 0, nullptr, out_f);
  }
}
